// sLSTMBlock_28226525070211
// MI455X (gfx1250) — matrix operand order repaired and checked by reading, not compiled, not run on hardware
//
#include <hip/hip_runtime.h>
#include <hip/hip_bf16.h>

typedef __attribute__((ext_vector_type(16))) __bf16 v16bf;
typedef __attribute__((ext_vector_type(8)))  float  v8f;

#define H_DIM 256
#define G_DIM 1024      // 4*H
#define M_DIM 341
#define M_PAD 352       // 341 padded to multiple of 32 (K for down proj, N for gate/up)
#define B_DIM 64
#define T_DIM 2048
#define NROWS 131072    // B*T

// ---------------------------------------------------------------------------
// WMMA helpers (bf16 A/B, f32 accumulate). Layouts per CDNA5 ISA 7.12.2.
// ---------------------------------------------------------------------------
__device__ __forceinline__ v8f wmma_bf16(v16bf a, v16bf b, v8f c) {
    return __builtin_amdgcn_wmma_f32_16x16x32_bf16(false, a, false, b,
                                                   (short)0, c, false, false);
}

// A fragment: 16x32 tile, row-major source. lane<16 -> rows 0..15 K {0..7,16..23};
// lanes 16..31 -> same rows, K shifted by +8.
__device__ __forceinline__ v16bf load_a_frag(const __bf16* base, long rowStride,
                                             int k0, int lane) {
    int half = lane >> 4, r = lane & 15;
    const __bf16* p = base + (long)r * rowStride + k0 + half * 8;
    v16bf a;
#pragma unroll
    for (int e = 0; e < 16; ++e) {
        int K = (e < 8) ? e : (e + 8);
        a[e] = p[K];
    }
    return a;
}

// B fragment: 32x16 tile of B = Wt^T where Wt is (Ntotal x K) row-major.
// lane column n = lane&15; lane<16 holds K {0..7,16..23}, lanes 16..31 K shifted by +8.
__device__ __forceinline__ v16bf load_b_frag(const __bf16* Wt, int ldk,
                                             int n0, int k0, int lane) {
    int half = lane >> 4, n = lane & 15;
    const __bf16* p = Wt + (long)(n0 + n) * ldk + k0 + half * 8;
    v16bf b;
#pragma unroll
    for (int e = 0; e < 16; ++e) b[e] = p[(e < 8) ? e : (e + 8)];
    return b;
}

__device__ __forceinline__ float gelu_tanh(float x) {
    float x3 = x * x * x;
    return 0.5f * x * (1.f + tanhf(0.7978845608028654f * (x + 0.044715f * x3)));
}

// ---------------------------------------------------------------------------
// CDNA5 async LDS fill (ASYNCcnt-tracked): per-lane 16B global -> LDS.
// ---------------------------------------------------------------------------
__device__ __forceinline__ void async_load_b128(unsigned lds_addr,
                                                unsigned long long gaddr) {
    asm volatile("global_load_async_to_lds_b128 %0, %1, off"
                 :: "v"(lds_addr), "v"(gaddr) : "memory");
}

__device__ __forceinline__ void wait_async0() {
#if __has_builtin(__builtin_amdgcn_s_wait_asynccnt)
    __builtin_amdgcn_s_wait_asynccnt(0);
#else
    asm volatile("s_wait_asynccnt 0x0" ::: "memory");
#endif
}

// ---------------------------------------------------------------------------
// Weight conversion fp32 -> bf16 with zero padding
// ---------------------------------------------------------------------------
__global__ void pad_convert_bf16(const float* __restrict__ src, __bf16* __restrict__ dst,
                                 int srows, int scols, int drows, int dcols) {
    int idx = blockIdx.x * blockDim.x + threadIdx.x;
    int total = drows * dcols;
    if (idx >= total) return;
    int r = idx / dcols, c = idx % dcols;
    float v = (r < srows && c < scols) ? src[r * scols + c] : 0.f;
    dst[idx] = (__bf16)v;
}

// ---------------------------------------------------------------------------
// LayerNorm1: one wave per 256-wide row -> bf16 output. b128 loads.
// ---------------------------------------------------------------------------
__global__ void __launch_bounds__(256) ln1_kernel(const float* __restrict__ x,
                                                  const float* __restrict__ w,
                                                  const float* __restrict__ b,
                                                  __bf16* __restrict__ nx) {
    int lane = threadIdx.x & 31, wave = threadIdx.x >> 5;
    long row = (long)blockIdx.x * 8 + wave;
    const float4* xr4 = (const float4*)(x + row * H_DIM);
    float4 a0 = xr4[lane];
    float4 a1 = xr4[lane + 32];
    float v[8] = {a0.x, a0.y, a0.z, a0.w, a1.x, a1.y, a1.z, a1.w};
    float s = 0.f, s2 = 0.f;
#pragma unroll
    for (int j = 0; j < 8; ++j) { s += v[j]; s2 += v[j] * v[j]; }
#pragma unroll
    for (int m = 16; m >= 1; m >>= 1) {
        s  += __shfl_xor(s,  m, 32);
        s2 += __shfl_xor(s2, m, 32);
    }
    float mu  = s * (1.f / H_DIM);
    float var = s2 * (1.f / H_DIM) - mu * mu;
    float inv = rsqrtf(var + 1e-5f);
#pragma unroll
    for (int j = 0; j < 8; ++j) {
        int c = (j < 4 ? lane * 4 + j : 128 + lane * 4 + (j - 4));
        nx[row * H_DIM + c] = (__bf16)((v[j] - mu) * inv * w[c] + b[c]);
    }
}

// ---------------------------------------------------------------------------
// sLSTM recurrence. 4 workgroups, each owns 16 batch rows (== WMMA M tile).
// Per step: pre = nx_t @ W^T + b + h @ R^T via WMMA, then pointwise gates.
// State (c,n,m,h) lives in LDS. x_t tiles are double-buffered via
// GLOBAL_LOAD_ASYNC_TO_LDS_B128: step t+1 streams in while step t computes.
// Dynamic LDS: pre(64K) + c/n/m(48K) + h(8K) + 2x nx(16K) = 136KB (< 320KB WGP)
// ---------------------------------------------------------------------------
__global__ void __launch_bounds__(256) slstm_kernel(const __bf16* __restrict__ nx,
                                                    const __bf16* __restrict__ Wbf,
                                                    const __bf16* __restrict__ Rbf,
                                                    const float* __restrict__ bias,
                                                    __bf16* __restrict__ hs) {
    extern __shared__ char smem[];
    float*  pre_sm = (float*)smem;                 // 16 x 1024
    float*  c_sm   = pre_sm + 16 * G_DIM;          // 16 x 256
    float*  n_sm   = c_sm + 16 * H_DIM;
    float*  m_sm   = n_sm + 16 * H_DIM;
    __bf16* h_sm   = (__bf16*)(m_sm + 16 * H_DIM); // 16 x 256
    __bf16* nx_sm0 = h_sm + 16 * H_DIM;            // 16 x 256 (buffer 0)
    __bf16* nx_sm1 = nx_sm0 + 16 * H_DIM;          // 16 x 256 (buffer 1)

    const int tid = threadIdx.x, lane = tid & 31, wave = tid >> 5;
    const int rb = blockIdx.x * 16;   // first batch row of this group

    for (int i = tid; i < 16 * H_DIM; i += 256) {
        c_sm[i] = 0.f; n_sm[i] = 0.f; m_sm[i] = 0.f; h_sm[i] = (__bf16)0.f;
    }

    // async-stage x_t tile: wave w fills rows 2w, 2w+1; each lane moves 16B.
    auto stage = [&](int t, __bf16* buf) {
#pragma unroll
        for (int rr = 0; rr < 2; ++rr) {
            int r = wave * 2 + rr;
            const __bf16* g = nx + (((long)(rb + r)) * T_DIM + t) * H_DIM;
            unsigned long long ga = (unsigned long long)(uintptr_t)g
                                    + (unsigned)(lane * 16);
            unsigned la = (unsigned)(uintptr_t)(buf + r * H_DIM) + lane * 16;
            async_load_b128(la, ga);
        }
    };

    stage(0, nx_sm0);

    for (int t = 0; t < T_DIM; ++t) {
        __bf16* nx_cur = (t & 1) ? nx_sm1 : nx_sm0;
        __bf16* nx_nxt = (t & 1) ? nx_sm0 : nx_sm1;

        wait_async0();        // this wave's tile rows are in LDS
        __syncthreads();      // everyone's rows are in LDS
        if (t + 1 < T_DIM) stage(t + 1, nx_nxt);   // overlap with GEMM below

        // GEMM: wave w computes pre columns [w*128, w*128+128)
#pragma unroll 1
        for (int nt = 0; nt < 8; ++nt) {
            int n0 = wave * 128 + nt * 16;
            float bb = bias[n0 + (lane & 15)];
            v8f acc;
#pragma unroll
            for (int r = 0; r < 8; ++r) acc[r] = bb;
#pragma unroll
            for (int kk = 0; kk < 8; ++kk) {
                v16bf a  = load_a_frag(nx_cur, H_DIM, kk * 32, lane);
                v16bf bm = load_b_frag(Wbf, H_DIM, n0, kk * 32, lane);
                acc = wmma_bf16(a, bm, acc);
            }
#pragma unroll
            for (int kk = 0; kk < 8; ++kk) {
                v16bf a  = load_a_frag(h_sm, H_DIM, kk * 32, lane);
                v16bf bm = load_b_frag(Rbf, H_DIM, n0, kk * 32, lane);
                acc = wmma_bf16(a, bm, acc);
            }
            int col = lane & 15, hh = lane >> 4;
#pragma unroll
            for (int r = 0; r < 8; ++r)
                pre_sm[(r + 8 * hh) * G_DIM + n0 + col] = acc[r];
        }
        __syncthreads();

        // pointwise gate update: thread -> row tid/16, 16 cols strided by 16
        int row = tid >> 4;
#pragma unroll 4
        for (int j = 0; j < 16; ++j) {
            int col = (tid & 15) + 16 * j;
            int idx = row * H_DIM + col;
            float z  = tanhf(pre_sm[row * G_DIM + col]);
            float li = pre_sm[row * G_DIM + H_DIM + col];
            float lf = pre_sm[row * G_DIM + 2 * H_DIM + col];
            float o  = 1.f / (1.f + expf(-pre_sm[row * G_DIM + 3 * H_DIM + col]));
            float mo = m_sm[idx];
            float mn = fmaxf(lf + mo, li);
            float ig = expf(li - mn);
            float fg = expf(lf + mo - mn);
            float c  = fg * c_sm[idx] + ig * z;
            float n  = fg * n_sm[idx] + ig;
            float h  = o * (c / n);
            c_sm[idx] = c; n_sm[idx] = n; m_sm[idx] = mn;
            h_sm[idx] = (__bf16)h;
            hs[(((long)(rb + row)) * T_DIM + t) * H_DIM + col] = (__bf16)h;
        }
        __syncthreads();
    }
}

// ---------------------------------------------------------------------------
// Fused gate/up projections: P = gelu(h@gateW^T + gb) * (h@upW^T + ub), bf16,
// N padded to 352 (pad columns come out exactly 0 from zero-padded weights).
// One wave per 16x16 tile pair, A-fragments shared between the two GEMMs.
// ---------------------------------------------------------------------------
__global__ void __launch_bounds__(256) gateup_kernel(const __bf16* __restrict__ hs,
                                                     const __bf16* __restrict__ gW,
                                                     const __bf16* __restrict__ uW,
                                                     const float* __restrict__ gb,
                                                     const float* __restrict__ ub,
                                                     __bf16* __restrict__ P) {
    const int lane = threadIdx.x & 31, wave = threadIdx.x >> 5;
    long gidx = (long)blockIdx.x * 8 + wave;
    long rowtile = gidx / (M_PAD / 16);
    int  ntile   = (int)(gidx % (M_PAD / 16));
    if (rowtile >= NROWS / 16) return;
    long row0 = rowtile * 16;
    int  n0   = ntile * 16;

    v8f accg, accu;
#pragma unroll
    for (int r = 0; r < 8; ++r) { accg[r] = 0.f; accu[r] = 0.f; }
#pragma unroll
    for (int kk = 0; kk < 8; ++kk) {
        v16bf a  = load_a_frag(hs + row0 * H_DIM, H_DIM, kk * 32, lane);
        v16bf bg = load_b_frag(gW, H_DIM, n0, kk * 32, lane);
        v16bf bu = load_b_frag(uW, H_DIM, n0, kk * 32, lane);
        accg = wmma_bf16(a, bg, accg);
        accu = wmma_bf16(a, bu, accu);
    }
    int col = n0 + (lane & 15), hh = lane >> 4;
    float gbias = (col < M_DIM) ? gb[col] : 0.f;
    float ubias = (col < M_DIM) ? ub[col] : 0.f;
#pragma unroll
    for (int r = 0; r < 8; ++r) {
        float g = gelu_tanh(accg[r] + gbias);
        float u = accu[r] + ubias;
        P[(row0 + r + 8 * hh) * M_PAD + col] = (__bf16)(g * u);
    }
}

// ---------------------------------------------------------------------------
// Fused down projection + LayerNorm2 + residual. One workgroup per 16 rows;
// full 256-wide row assembled in LDS for the LN reduction.
// ---------------------------------------------------------------------------
__global__ void __launch_bounds__(256) down_ln_kernel(const __bf16* __restrict__ P,
                                                      const __bf16* __restrict__ dW,
                                                      const float* __restrict__ db,
                                                      const float* __restrict__ ln2w,
                                                      const float* __restrict__ ln2b,
                                                      const float* __restrict__ x,
                                                      float* __restrict__ out) {
    __shared__ float osm[16 * H_DIM];
    const int tid = threadIdx.x, lane = tid & 31, wave = tid >> 5;
    long row0 = (long)blockIdx.x * 16;

#pragma unroll 1
    for (int sub = 0; sub < 2; ++sub) {
        int n0 = wave * 32 + sub * 16;
        v8f acc;
#pragma unroll
        for (int r = 0; r < 8; ++r) acc[r] = 0.f;
#pragma unroll
        for (int kk = 0; kk < M_PAD / 32; ++kk) {
            v16bf a  = load_a_frag(P + row0 * M_PAD, M_PAD, kk * 32, lane);
            v16bf bm = load_b_frag(dW, M_PAD, n0, kk * 32, lane);
            acc = wmma_bf16(a, bm, acc);
        }
        int col = n0 + (lane & 15), hh = lane >> 4;
        float bb = db[col];
#pragma unroll
        for (int r = 0; r < 8; ++r)
            osm[(r + 8 * hh) * H_DIM + col] = acc[r] + bb;
    }
    __syncthreads();

    int row = tid >> 4, cbase = tid & 15;
    float v[16], s = 0.f, s2 = 0.f;
#pragma unroll
    for (int j = 0; j < 16; ++j) {
        v[j] = osm[row * H_DIM + cbase + 16 * j];
        s += v[j]; s2 += v[j] * v[j];
    }
#pragma unroll
    for (int m = 8; m >= 1; m >>= 1) {   // 16 threads of a row sit in one 16-lane half
        s  += __shfl_xor(s,  m, 16);
        s2 += __shfl_xor(s2, m, 16);
    }
    float mu  = s * (1.f / H_DIM);
    float var = s2 * (1.f / H_DIM) - mu * mu;
    float inv = rsqrtf(var + 1e-5f);
    long grow = row0 + row;
#pragma unroll
    for (int j = 0; j < 16; ++j) {
        int c = cbase + 16 * j;
        out[grow * H_DIM + c] =
            (v[j] - mu) * inv * ln2w[c] + ln2b[c] + x[grow * H_DIM + c];
    }
}

// ---------------------------------------------------------------------------
extern "C" void kernel_launch(void* const* d_in, const int* in_sizes, int n_in,
                              void* d_out, int out_size, void* d_ws, size_t ws_size,
                              hipStream_t stream) {
    const float* x     = (const float*)d_in[0];
    const float* ln1w  = (const float*)d_in[1];
    const float* ln1b  = (const float*)d_in[2];
    const float* W     = (const float*)d_in[3];
    const float* R     = (const float*)d_in[4];
    const float* bvec  = (const float*)d_in[5];
    const float* upW   = (const float*)d_in[6];
    const float* upb   = (const float*)d_in[7];
    const float* gateW = (const float*)d_in[8];
    const float* gateb = (const float*)d_in[9];
    const float* downW = (const float*)d_in[10];
    const float* downb = (const float*)d_in[11];
    const float* ln2w  = (const float*)d_in[12];
    const float* ln2b  = (const float*)d_in[13];
    float* out = (float*)d_out;

    char* p = (char*)d_ws;
    auto carve = [&](size_t bytes) {
        void* r = (void*)p;
        p += (bytes + 255) & ~(size_t)255;
        return r;
    };
    __bf16* nx  = (__bf16*)carve((size_t)NROWS * H_DIM * 2);
    __bf16* hsb = (__bf16*)carve((size_t)NROWS * H_DIM * 2);
    __bf16* P   = (__bf16*)carve((size_t)NROWS * M_PAD * 2);
    __bf16* Wbf = (__bf16*)carve((size_t)G_DIM * H_DIM * 2);
    __bf16* Rbf = (__bf16*)carve((size_t)G_DIM * H_DIM * 2);
    __bf16* gWb = (__bf16*)carve((size_t)M_PAD * H_DIM * 2);
    __bf16* uWb = (__bf16*)carve((size_t)M_PAD * H_DIM * 2);
    __bf16* dWb = (__bf16*)carve((size_t)H_DIM * M_PAD * 2);

    // weight conversion / padding
    pad_convert_bf16<<<(G_DIM * H_DIM + 255) / 256, 256, 0, stream>>>(W, Wbf, G_DIM, H_DIM, G_DIM, H_DIM);
    pad_convert_bf16<<<(G_DIM * H_DIM + 255) / 256, 256, 0, stream>>>(R, Rbf, G_DIM, H_DIM, G_DIM, H_DIM);
    pad_convert_bf16<<<(M_PAD * H_DIM + 255) / 256, 256, 0, stream>>>(gateW, gWb, M_DIM, H_DIM, M_PAD, H_DIM);
    pad_convert_bf16<<<(M_PAD * H_DIM + 255) / 256, 256, 0, stream>>>(upW, uWb, M_DIM, H_DIM, M_PAD, H_DIM);
    pad_convert_bf16<<<(H_DIM * M_PAD + 255) / 256, 256, 0, stream>>>(downW, dWb, H_DIM, M_DIM, H_DIM, M_PAD);

    // LN1 -> bf16 activations
    ln1_kernel<<<NROWS / 8, 256, 0, stream>>>(x, ln1w, ln1b, nx);

    // sequential sLSTM scan: 4 persistent workgroups (16 batch rows each)
    size_t slds = 16 * G_DIM * 4 + 3 * 16 * H_DIM * 4 + 3 * 16 * H_DIM * 2; // 136 KB
    hipFuncSetAttribute((const void*)slstm_kernel,
                        hipFuncAttributeMaxDynamicSharedMemorySize, (int)slds);
    slstm_kernel<<<4, 256, slds, stream>>>(nx, Wbf, Rbf, bvec, hsb);

    // fused gate/up + gelu*mul
    gateup_kernel<<<(NROWS / 16) * (M_PAD / 16) / 8, 256, 0, stream>>>(hsb, gWb, uWb, gateb, upb, P);

    // fused down proj + LN2 + residual
    down_ln_kernel<<<NROWS / 16, 256, 0, stream>>>(P, dWb, downb, ln2w, ln2b, x, out);
}
